// CIFARViT_61546881352307
// MI455X (gfx1250) — matrix operand order repaired and checked by reading, not compiled, not run on hardware
//
#include <hip/hip_runtime.h>
#include <hip/hip_bf16.h>
#include <math.h>

// ---------------------------------------------------------------------------
// CDNA5 (gfx1250) spectral-ViT forward.
// Large GEMMs: fragment-packed bf16 operands, 8-wave workgroup computing a
// 128x64 block with LDS-staged B tiles (8x weight-reuse) + v_wmma bf16.
// Eigen subspace iteration: v_wmma_f32_16x16x4_f32 (full f32 precision).
// wave32 throughout.
// ---------------------------------------------------------------------------

typedef __attribute__((ext_vector_type(16))) __bf16 v16bf;
typedef __attribute__((ext_vector_type(8)))  float  v8f;
typedef __attribute__((ext_vector_type(2)))  float  v2f;

#define D_LAYERS 8
#define CDIM     768
#define HHEADS   8
#define BATCH    32
#define NTOK     256
#define CH       96      // C/H
#define NEIG     5
#define XS       17      // LDS row stride for eigen blocks (bank-conflict pad)
#define POWER_ITERS 30

#if __has_builtin(__builtin_amdgcn_wmma_f32_16x16x4_f32)
#define HAVE_WMMA_F32_K4 1
#endif

// round-to-nearest-even float -> bf16 (bit twiddle; used only in pack kernels)
static __device__ __forceinline__ __bf16 f2bf(float x) {
    union { float f; unsigned u; } v; v.f = x;
    unsigned r = (v.u + 0x7FFFu + ((v.u >> 16) & 1u)) >> 16;
    union { unsigned short s; __bf16 b; } o; o.s = (unsigned short)r;
    return o.b;
}

// A-fragment K index for 16-bit 16x32 A tiles (ISA 7.12.2)
static __device__ __forceinline__ int ka_idx(int e, int hf) {
    int j = e >> 1, pos = e & 1;
    return ((j & 4) ? 16 : 0) + hf * 8 + ((j & 3) << 1) + pos;
}
// B-fragment K index for 16-bit 32x16 B tiles: same K order as A fragments
static __device__ __forceinline__ int kb_idx(int e, int hf) {
    return ((e & 8) ? 16 : 0) + hf * 8 + (e & 7);
}

// ---------------- block reductions (blockDim.x == 256) ----------------------
static __device__ __forceinline__ float block_sum(float v, float* red) {
    int t = threadIdx.x;
    red[t] = v; __syncthreads();
    for (int s = 128; s > 0; s >>= 1) { if (t < s) red[t] += red[t + s]; __syncthreads(); }
    float r = red[0]; __syncthreads();
    return r;
}
static __device__ __forceinline__ float block_max(float v, float* red) {
    int t = threadIdx.x;
    red[t] = v; __syncthreads();
    for (int s = 128; s > 0; s >>= 1) { if (t < s) red[t] = fmaxf(red[t], red[t + s]); __syncthreads(); }
    float r = red[0]; __syncthreads();
    return r;
}

// ---------------------------------------------------------------------------
// Operand packing: f32 matrix -> fragment-major bf16.
// Batch offset: base + (z/zdiv)*strideHi + (z%zdiv)*strideLo  (element units)
// ---------------------------------------------------------------------------
__global__ __launch_bounds__(256) void pack_a_bf16(
    const float* __restrict__ in, int lda, long long strideHi, long long strideLo,
    int zdiv, __bf16* __restrict__ out, long long outStride, int Ktiles)
{
    int tid = threadIdx.x;
    int tile = blockIdx.x * 8 + (tid >> 5);
    int lane = tid & 31, m = lane & 15, hf = lane >> 4;
    int rt = tile / Ktiles, kt = tile - rt * Ktiles;
    int z = blockIdx.z;
    const float* src = in + (size_t)(z / zdiv) * strideHi + (size_t)(z % zdiv) * strideLo
                          + (size_t)(rt * 16 + m) * lda + kt * 32;
    __bf16* dst = out + (size_t)z * outStride + (size_t)tile * 512 + lane * 16;
    #pragma unroll
    for (int e = 0; e < 16; ++e) dst[e] = f2bf(src[ka_idx(e, hf)]);
}

__global__ __launch_bounds__(256) void pack_b_bf16(
    const float* __restrict__ in, int ldw, long long strideHi, long long strideLo,
    int zdiv, __bf16* __restrict__ out, long long outStride, int Ntiles)
{
    int tid = threadIdx.x;
    int tile = blockIdx.x * 8 + (tid >> 5);
    int lane = tid & 31, m = lane & 15, hf = lane >> 4;
    int kt = tile / Ntiles, nt = tile - kt * Ntiles;
    int z = blockIdx.z;
    const float* base = in + (size_t)(z / zdiv) * strideHi + (size_t)(z % zdiv) * strideLo;
    __bf16* dst = out + (size_t)z * outStride + (size_t)tile * 512 + lane * 16;
    #pragma unroll
    for (int e = 0; e < 16; ++e)
        dst[e] = f2bf(base[(size_t)(kt * 32 + kb_idx(e, hf)) * ldw + nt * 16 + m]);
}

// ---------------------------------------------------------------------------
// Patch embed: 2x2/s2 conv (=12-tap GEMM) + bias + pos_emb. h[b,n,c].
// ---------------------------------------------------------------------------
__global__ __launch_bounds__(256) void patch_embed_kernel(
    const float* __restrict__ x, const float* __restrict__ pw,
    const float* __restrict__ pb, const float* __restrict__ pos,
    float* __restrict__ h)
{
    int bn = blockIdx.x;
    int b = bn >> 8, n = bn & 255;
    int ph = n >> 4, pwi = n & 15;
    __shared__ float pix[12];
    int t = threadIdx.x;
    if (t < 12) {
        int ci = t >> 2, kh = (t >> 1) & 1, kw = t & 1;
        pix[t] = x[((b * 3 + ci) * 32 + (2 * ph + kh)) * 32 + (2 * pwi + kw)];
    }
    __syncthreads();
    #pragma unroll
    for (int j = 0; j < 3; ++j) {
        int c = t + j * 256;
        float s = pb[c];
        #pragma unroll
        for (int q = 0; q < 12; ++q) {
            int ci = q >> 2, kh = (q >> 1) & 1, kw = q & 1;
            s += pix[q] * pw[((c * 3 + ci) * 2 + kh) * 2 + kw];
        }
        h[((size_t)b * NTOK + n) * CDIM + c] = s + pos[n * CDIM + c];
    }
}

// ---------------------------------------------------------------------------
// Packed-operand GEMM: C[M,N] = act(A @ W + bias) + resid.
// 256 threads = 8 waves; block tile 128x64. Per k-step the 4 B tiles (4KB)
// are staged once through LDS and consumed by all 8 waves (8x B reuse);
// the next k-step's B ride in registers while WMMAs run (double buffering).
// ---------------------------------------------------------------------------
__global__ __launch_bounds__(256) void gemm_packed_wmma(
    const __bf16* __restrict__ Ap, const __bf16* __restrict__ Bp,
    const float* __restrict__ bias, const float* __restrict__ resid,
    float* __restrict__ C, int ldc, int Ktiles, int Ntiles, int act)
{
    __shared__ __align__(16) __bf16 Bs[4 * 512];   // 4 B tiles = 4KB
    int t = threadIdx.x;
    int wave = t >> 5, lane = t & 31, m = lane & 15, hf = lane >> 4;
    int rt = blockIdx.x * 8 + wave;                // row tile owned by this wave
    int ctg = blockIdx.y;                          // group of 4 col tiles

    v8f acc[4];
    #pragma unroll
    for (int j = 0; j < 4; ++j) acc[j] = (v8f){0.f,0.f,0.f,0.f,0.f,0.f,0.f,0.f};

    const __bf16* a_ptr = Ap + ((size_t)rt * Ktiles) * 512 + lane * 16;
    const __bf16* b_ptr = Bp + ((size_t)(ctg * 4)) * 512 + t * 8;   // 16B per thread
    const size_t  b_step = (size_t)Ntiles * 512;

    float4 nxt = *(const float4*)b_ptr;            // stage k-step 0
    for (int kt = 0; kt < Ktiles; ++kt) {
        float4 cur = nxt;
        if (kt + 1 < Ktiles)                       // global load overlaps compute
            nxt = *(const float4*)(b_ptr + (size_t)(kt + 1) * b_step);
        __syncthreads();                           // previous compute done
        *(float4*)&Bs[t * 8] = cur;                // ds_store_b128
        __syncthreads();                           // B tiles visible

        __builtin_prefetch(a_ptr + 512, 0, 3);     // pull next A frag toward WGP
        v16bf af = *(const v16bf*)a_ptr;  a_ptr += 512;
        v16bf b0 = *(const v16bf*)(Bs + 0 * 512 + lane * 16);
        v16bf b1 = *(const v16bf*)(Bs + 1 * 512 + lane * 16);
        v16bf b2 = *(const v16bf*)(Bs + 2 * 512 + lane * 16);
        v16bf b3 = *(const v16bf*)(Bs + 3 * 512 + lane * 16);
        acc[0] = __builtin_amdgcn_wmma_f32_16x16x32_bf16(false, af, false, b0, (short)0, acc[0], false, false);
        acc[1] = __builtin_amdgcn_wmma_f32_16x16x32_bf16(false, af, false, b1, (short)0, acc[1], false, false);
        acc[2] = __builtin_amdgcn_wmma_f32_16x16x32_bf16(false, af, false, b2, (short)0, acc[2], false, false);
        acc[3] = __builtin_amdgcn_wmma_f32_16x16x32_bf16(false, af, false, b3, (short)0, acc[3], false, false);
    }
    int row0 = rt * 16, col0 = ctg * 64;
    #pragma unroll
    for (int e = 0; e < 8; ++e) {
        int r = row0 + hf * 8 + e;
        #pragma unroll
        for (int j = 0; j < 4; ++j) {
            int cidx = col0 + j * 16 + m;
            float v = acc[j][e];
            if (bias)  v += bias[cidx];
            if (act == 1) v = 0.5f * v * (1.0f + erff(v * 0.70710678118f)); // exact GELU
            if (resid) v += resid[(size_t)r * ldc + cidx];
            C[(size_t)r * ldc + cidx] = v;
        }
    }
}

// ---------------------------------------------------------------------------
// Attention scores from packed q: S[b,h] = c^-0.5 * q @ q^T. K=96 (3 k-tiles).
// ---------------------------------------------------------------------------
__global__ __launch_bounds__(32) void attn_scores_packed(
    const __bf16* __restrict__ qA, const __bf16* __restrict__ qB,
    float* __restrict__ S)
{
    int bh = blockIdx.z;
    int rt = blockIdx.x, nt = blockIdx.y;
    int lane = threadIdx.x & 31;
    int m = lane & 15, hf = lane >> 4;
    const __bf16* a = qA + (size_t)bh * (16 * 3 * 512) + (size_t)(rt * 3) * 512 + lane * 16;
    const __bf16* b = qB + (size_t)bh * (3 * 16 * 512) + (size_t)nt * 512 + lane * 16;

    v8f acc = (v8f){0.f,0.f,0.f,0.f,0.f,0.f,0.f,0.f};
    #pragma unroll
    for (int kt = 0; kt < 3; ++kt) {
        v16bf af = *(const v16bf*)(a + (size_t)kt * 512);
        v16bf bf = *(const v16bf*)(b + (size_t)kt * 16 * 512);
        acc = __builtin_amdgcn_wmma_f32_16x16x32_bf16(false, af, false, bf, (short)0, acc, false, false);
    }
    const float scale = 0.10206207262f; // 96^-0.5
    #pragma unroll
    for (int e = 0; e < 8; ++e) {
        int r = rt * 16 + hf * 8 + e;
        S[((size_t)bh * NTOK + r) * NTOK + nt * 16 + m] = acc[e] * scale;
    }
}

// ---------------------------------------------------------------------------
// Row softmax with masked diagonal (in place). One 256-thread block per row.
// ---------------------------------------------------------------------------
__global__ __launch_bounds__(256) void softmax_mask_diag(float* __restrict__ S)
{
    __shared__ float red[256];
    int rowg = blockIdx.x;
    int r = rowg & 255;
    float* row = S + (size_t)rowg * NTOK;
    int t = threadIdx.x;
    float v = row[t];
    if (t == r) v = -INFINITY;
    float mx = block_max(v, red);
    float ex = __expf(v - mx);
    float sm = block_sum(ex, red);
    row[t] = ex / sm;
}

// ---------------------------------------------------------------------------
// Symmetrize + normalized Laplacian. rowsum(P)==1 exactly.
// ---------------------------------------------------------------------------
__global__ __launch_bounds__(256) void laplacian_kernel(
    const float* __restrict__ P, float* __restrict__ Lout)
{
    int bh = blockIdx.x; int t = threadIdx.x;
    const float* Pb = P + (size_t)bh * NTOK * NTOK;
    float* Lb = Lout + (size_t)bh * NTOK * NTOK;
    __shared__ float degs[256]; __shared__ float invs[256];
    float cs = 0.f;
    for (int mr = 0; mr < NTOK; ++mr) cs += Pb[mr * NTOK + t];
    float deg = 0.5f * (1.0f + cs);
    degs[t] = deg; invs[t] = rsqrtf(deg);
    __syncthreads();
    for (int idx = t; idx < NTOK * NTOK; idx += 256) {
        int mr = idx >> 8, n = idx & 255;
        float asym = 0.5f * (Pb[idx] + Pb[n * NTOK + mr]);
        float val = ((mr == n) ? degs[mr] : 0.f) - asym;
        Lb[idx] = val * invs[mr] * invs[n];
    }
}

// ---------------------------------------------------------------------------
// Eigen solver: subspace iteration on (2I - L), WMMA f32 16x16x4,
// MGS re-orthonormalization, 6x6 Rayleigh-Ritz. One block per (b,h).
// ---------------------------------------------------------------------------
static __device__ __forceinline__ void lmul_block(
    const float* __restrict__ L, const float* Xs, float* Ys, float twoI)
{
    int t = threadIdx.x;
    int wave = t >> 5, lane = t & 31, m = lane & 15, hf = lane >> 4;
    for (int rt = wave; rt < 16; rt += 8) {
        int row0 = rt * 16;
        v8f acc = (v8f){0.f,0.f,0.f,0.f,0.f,0.f,0.f,0.f};
#ifdef HAVE_WMMA_F32_K4
        const float* Lr = L + (size_t)(row0 + m) * NTOK;
        for (int k0 = 0; k0 < NTOK; k0 += 4) {
            v2f af; af.x = Lr[k0 + hf * 2]; af.y = Lr[k0 + hf * 2 + 1];
            v2f bf; bf.x = Xs[(k0 + hf * 2) * XS + m]; bf.y = Xs[(k0 + hf * 2 + 1) * XS + m];
            acc = __builtin_amdgcn_wmma_f32_16x16x4_f32(
                false, af, false, bf, (short)0, acc, false, false);
        }
#else
        #pragma unroll
        for (int e = 0; e < 8; ++e) {
            int r = row0 + hf * 8 + e;
            const float* Lrr = L + (size_t)r * NTOK;
            float s = 0.f;
            for (int k = 0; k < NTOK; ++k) s += Lrr[k] * Xs[k * XS + m];
            acc[e] = s;
        }
#endif
        #pragma unroll
        for (int e = 0; e < 8; ++e) {
            int r = row0 + hf * 8 + e;
            Ys[r * XS + m] = twoI * Xs[r * XS + m] - acc[e];
        }
    }
    __syncthreads();
}

static __device__ __forceinline__ void mgs6(float* Xs, float* Ys, float* red)
{
    int t = threadIdx.x;
    for (int j = 0; j < 6; ++j) {
        float y = Ys[t * XS + j];
        for (int i = 0; i < j; ++i) {
            float x = Xs[t * XS + i];
            float r = block_sum(x * y, red);
            y -= r * x;
        }
        float nr = block_sum(y * y, red);
        float inv = rsqrtf(nr + 1e-18f);
        Xs[t * XS + j] = y * inv;
        __syncthreads();
    }
}

__global__ __launch_bounds__(256) void spectral_eigs_kernel(
    const float* __restrict__ Lap, float* __restrict__ eig)
{
    __shared__ float Xs[NTOK * XS];
    __shared__ float Ys[NTOK * XS];
    __shared__ float red[256];
    __shared__ float Ts[36];
    __shared__ float Vsel[6 * NEIG];
    int bh = blockIdx.x;
    const float* L = Lap + (size_t)bh * NTOK * NTOK;
    int t = threadIdx.x;

    Ys[t * XS + 0] = 1.0f;
    #pragma unroll
    for (int j = 1; j < 6; ++j)
        Ys[t * XS + j] = __sinf(0.37f * (float)(t + 1) * (float)(j + 1) + 0.11f * (float)j);
    #pragma unroll
    for (int j = 6; j < 16; ++j) { Ys[t * XS + j] = 0.f; Xs[t * XS + j] = 0.f; }
    __syncthreads();
    mgs6(Xs, Ys, red);

    for (int it = 0; it < POWER_ITERS; ++it) {
        lmul_block(L, Xs, Ys, 2.0f);   // Y = (2I - L) X
        mgs6(Xs, Ys, red);
    }

    lmul_block(L, Xs, Ys, 0.0f);       // Ys = -L X
    for (int i = 0; i < 6; ++i)
        for (int j = i; j < 6; ++j) {
            float d = block_sum(Xs[t * XS + i] * (-Ys[t * XS + j]), red);
            if (t == 0) { Ts[i * 6 + j] = d; Ts[j * 6 + i] = d; }
        }
    __syncthreads();

    if (t == 0) {
        float A6[36], V6[36];
        for (int i = 0; i < 36; ++i) { A6[i] = Ts[i]; V6[i] = 0.f; }
        for (int i = 0; i < 6; ++i) V6[i * 6 + i] = 1.f;
        for (int sw = 0; sw < 12; ++sw)
            for (int p = 0; p < 5; ++p)
                for (int q = p + 1; q < 6; ++q) {
                    float apq = A6[p * 6 + q];
                    if (fabsf(apq) < 1e-12f) continue;
                    float tau = (A6[q * 6 + q] - A6[p * 6 + p]) / (2.f * apq);
                    float tt = (tau >= 0.f ? 1.f : -1.f) / (fabsf(tau) + sqrtf(1.f + tau * tau));
                    float c = rsqrtf(1.f + tt * tt), s = tt * c;
                    for (int k = 0; k < 6; ++k) {
                        float akp = A6[k * 6 + p], akq = A6[k * 6 + q];
                        A6[k * 6 + p] = c * akp - s * akq;
                        A6[k * 6 + q] = s * akp + c * akq;
                    }
                    for (int k = 0; k < 6; ++k) {
                        float apk = A6[p * 6 + k], aqk = A6[q * 6 + k];
                        A6[p * 6 + k] = c * apk - s * aqk;
                        A6[q * 6 + k] = s * apk + c * aqk;
                    }
                    for (int k = 0; k < 6; ++k) {
                        float vkp = V6[k * 6 + p], vkq = V6[k * 6 + q];
                        V6[k * 6 + p] = c * vkp - s * vkq;
                        V6[k * 6 + q] = s * vkp + c * vkq;
                    }
                }
        int idx[6] = {0, 1, 2, 3, 4, 5};
        for (int i = 1; i < 6; ++i) {
            int key = idx[i]; float kv = A6[key * 6 + key]; int j = i - 1;
            while (j >= 0 && A6[idx[j] * 6 + idx[j]] > kv) { idx[j + 1] = idx[j]; --j; }
            idx[j + 1] = key;
        }
        for (int e = 0; e < NEIG; ++e)
            for (int j = 0; j < 6; ++j)
                Vsel[j * NEIG + e] = V6[j * 6 + idx[e + 1]]; // skip trivial mode
    }
    __syncthreads();

    #pragma unroll
    for (int e = 0; e < NEIG; ++e) {
        float s = 0.f;
        #pragma unroll
        for (int j = 0; j < 6; ++j) s += Xs[t * XS + j] * Vsel[j * NEIG + e];
        eig[((size_t)bh * NEIG + e) * NTOK + t] = s;
    }
}

// ---------------------------------------------------------------------------
// group logits = combs(32x5 of +-1) @ eigvecs(5x256), softmax over tokens.
// ---------------------------------------------------------------------------
__global__ __launch_bounds__(256) void group_attn_softmax(
    const float* __restrict__ eig, float* __restrict__ G)
{
    __shared__ float ev[NEIG][256];
    __shared__ float red[256];
    int bh = blockIdx.x; int t = threadIdx.x;
    #pragma unroll
    for (int e = 0; e < NEIG; ++e) ev[e][t] = eig[((size_t)bh * NEIG + e) * NTOK + t];
    __syncthreads();
    for (int o = 0; o < 32; ++o) {
        float lg = 0.f;
        #pragma unroll
        for (int e = 0; e < NEIG; ++e) {
            float sgn = ((o >> (4 - e)) & 1) ? -1.f : 1.f;
            lg += sgn * ev[e][t];
        }
        float mx = block_max(lg, red);
        float ex = __expf(lg - mx);
        float sm = block_sum(ex, red);
        G[((size_t)bh * 32 + o) * NTOK + t] = ex / sm;
    }
}

// ---------------------------------------------------------------------------
// out[b,h,o,:] = G @ v  -> attn_out[b, h*32+o, 96]; packed operands, K=256.
// ---------------------------------------------------------------------------
__global__ __launch_bounds__(32) void group_out_packed(
    const __bf16* __restrict__ gA, const __bf16* __restrict__ vB,
    float* __restrict__ aout)
{
    int bh = blockIdx.z; int b = bh >> 3, hh = bh & 7;
    int rt = blockIdx.x;             // o tile: 0..1
    int nt = blockIdx.y;             // c tile: 0..5
    int lane = threadIdx.x & 31;
    int m = lane & 15, hf = lane >> 4;
    const __bf16* a = gA + (size_t)bh * (2 * 8 * 512) + (size_t)(rt * 8) * 512 + lane * 16;
    const __bf16* v = vB + (size_t)bh * (8 * 6 * 512) + (size_t)nt * 512 + lane * 16;

    v8f acc = (v8f){0.f,0.f,0.f,0.f,0.f,0.f,0.f,0.f};
    #pragma unroll
    for (int kt = 0; kt < 8; ++kt) {
        v16bf af = *(const v16bf*)(a + (size_t)kt * 512);
        v16bf bf = *(const v16bf*)(v + (size_t)kt * 6 * 512);
        acc = __builtin_amdgcn_wmma_f32_16x16x32_bf16(false, af, false, bf, (short)0, acc, false, false);
    }
    #pragma unroll
    for (int e = 0; e < 8; ++e) {
        int o = rt * 16 + hf * 8 + e;
        int tok = hh * 32 + o;
        aout[((size_t)b * NTOK + tok) * CH + nt * 16 + m] = acc[e];
    }
}

// ---------------------------------------------------------------------------
// Row LayerNorm over C=768.
// ---------------------------------------------------------------------------
__global__ __launch_bounds__(256) void layernorm_kernel(
    const float* __restrict__ in, const float* __restrict__ sc,
    const float* __restrict__ bs, float* __restrict__ out)
{
    __shared__ float red[256];
    int row = blockIdx.x; int t = threadIdx.x;
    const float* r = in + (size_t)row * CDIM;
    float* o = out + (size_t)row * CDIM;
    float v0 = r[t], v1 = r[t + 256], v2 = r[t + 512];
    float mean = block_sum(v0 + v1 + v2, red) * (1.0f / CDIM);
    float d0 = v0 - mean, d1 = v1 - mean, d2 = v2 - mean;
    float var = block_sum(d0 * d0 + d1 * d1 + d2 * d2, red) * (1.0f / CDIM);
    float inv = rsqrtf(var + 1e-5f);
    o[t]       = d0 * inv * sc[t]       + bs[t];
    o[t + 256] = d1 * inv * sc[t + 256] + bs[t + 256];
    o[t + 512] = d2 * inv * sc[t + 512] + bs[t + 512];
}

// ---------------------------------------------------------------------------
// Final: mean over tokens, LN, 768->10 head.
// ---------------------------------------------------------------------------
__global__ __launch_bounds__(256) void head_kernel(
    const float* __restrict__ h, const float* __restrict__ ln_s,
    const float* __restrict__ ln_b, const float* __restrict__ hw,
    const float* __restrict__ hb, float* __restrict__ out)
{
    __shared__ float feat[CDIM];
    __shared__ float red[256];
    int b = blockIdx.x; int t = threadIdx.x;
    #pragma unroll
    for (int j = 0; j < 3; ++j) {
        int c = t + j * 256;
        float s = 0.f;
        for (int n = 0; n < NTOK; ++n) s += h[((size_t)b * NTOK + n) * CDIM + c];
        feat[c] = s * (1.0f / NTOK);
    }
    __syncthreads();
    float mean = block_sum(feat[t] + feat[t + 256] + feat[t + 512], red) * (1.0f / CDIM);
    float d0 = feat[t] - mean, d1 = feat[t + 256] - mean, d2 = feat[t + 512] - mean;
    float var = block_sum(d0 * d0 + d1 * d1 + d2 * d2, red) * (1.0f / CDIM);
    float inv = rsqrtf(var + 1e-5f);
    feat[t]       = d0 * inv * ln_s[t]       + ln_b[t];
    feat[t + 256] = d1 * inv * ln_s[t + 256] + ln_b[t + 256];
    feat[t + 512] = d2 * inv * ln_s[t + 512] + ln_b[t + 512];
    __syncthreads();
    if (t < 10) {
        float s = hb[t];
        for (int c = 0; c < CDIM; ++c) s += feat[c] * hw[c * 10 + t];
        out[b * 10 + t] = s;
    }
}

// ---------------------------------------------------------------------------
extern "C" void kernel_launch(void* const* d_in, const int* in_sizes, int n_in,
                              void* d_out, int out_size, void* d_ws, size_t ws_size,
                              hipStream_t stream)
{
    (void)in_sizes; (void)n_in; (void)out_size; (void)ws_size;
    const float* x       = (const float*)d_in[0];
    const float* patch_w = (const float*)d_in[1];
    const float* patch_b = (const float*)d_in[2];
    const float* pos_emb = (const float*)d_in[3];
    const float* qkv_w   = (const float*)d_in[4];
    const float* proj_w  = (const float*)d_in[5];
    const float* proj_b  = (const float*)d_in[6];
    const float* g_ln_s  = (const float*)d_in[7];
    const float* g_ln_b  = (const float*)d_in[8];
    const float* mlp_w1  = (const float*)d_in[9];
    const float* mlp_b1  = (const float*)d_in[10];
    const float* mlp_w2  = (const float*)d_in[11];
    const float* mlp_b2  = (const float*)d_in[12];
    const float* ln_s    = (const float*)d_in[13];
    const float* ln_b    = (const float*)d_in[14];
    const float* head_w  = (const float*)d_in[15];
    const float* head_b  = (const float*)d_in[16];
    float* out = (float*)d_out;
    float* ws  = (float*)d_ws;

    const size_t MROWS = (size_t)BATCH * NTOK;  // 8192
    const int BH = BATCH * HHEADS;              // 256

    // ---- f32 workspace ----
    size_t off = 0;
    float* h    = ws + off; off += MROWS * CDIM;
    float* qkv  = ws + off; off += MROWS * 3 * CDIM;
    float* S    = ws + off; off += (size_t)BH * NTOK * NTOK;
    float* lap  = ws + off; off += (size_t)BH * NTOK * NTOK;
    float* eig  = ws + off; off += (size_t)BH * NEIG * NTOK;
    float* G    = ws + off; off += (size_t)BH * 32 * NTOK;
    float* aout = ws + off; off += MROWS * CH;
    float* hn   = ws + off; off += MROWS * CDIM;
    float* mid  = ws + off; off += MROWS * 4 * CDIM;

    // ---- bf16 packed workspace ----
    __bf16* bws = (__bf16*)(ws + off);
    size_t bo = 0;
    __bf16* hA    = bws + bo; bo += MROWS * CDIM;
    __bf16* hnA   = bws + bo; bo += MROWS * CDIM;
    __bf16* midA  = bws + bo; bo += MROWS * 4 * CDIM;
    __bf16* aoutA = bws + bo; bo += MROWS * CH;
    __bf16* qA    = bws + bo; bo += (size_t)BH * NTOK * CH;
    __bf16* qB    = bws + bo; bo += (size_t)BH * NTOK * CH;
    __bf16* vB    = bws + bo; bo += (size_t)BH * NTOK * CH;
    __bf16* gAp   = bws + bo; bo += (size_t)BH * 32 * NTOK;
    __bf16* WqkvP = bws + bo; bo += (size_t)D_LAYERS * CDIM * 3 * CDIM;
    __bf16* WprojP= bws + bo; bo += (size_t)D_LAYERS * CH * CDIM;
    __bf16* W1P   = bws + bo; bo += (size_t)D_LAYERS * CDIM * 4 * CDIM;
    __bf16* W2P   = bws + bo; bo += (size_t)D_LAYERS * 4 * CDIM * CDIM;

    // ---- pack all weights once ----
    pack_b_bf16<<<dim3(3456 / 8, 1, D_LAYERS), 256, 0, stream>>>(
        qkv_w, 3 * CDIM, (long long)CDIM * 3 * CDIM, 0, 1, WqkvP, (long long)CDIM * 3 * CDIM, 144);
    pack_b_bf16<<<dim3(144 / 8, 1, D_LAYERS), 256, 0, stream>>>(
        proj_w, CDIM, (long long)CH * CDIM, 0, 1, WprojP, (long long)CH * CDIM, 48);
    pack_b_bf16<<<dim3(4608 / 8, 1, D_LAYERS), 256, 0, stream>>>(
        mlp_w1, 4 * CDIM, (long long)CDIM * 4 * CDIM, 0, 1, W1P, (long long)CDIM * 4 * CDIM, 192);
    pack_b_bf16<<<dim3(4608 / 8, 1, D_LAYERS), 256, 0, stream>>>(
        mlp_w2, CDIM, (long long)4 * CDIM * CDIM, 0, 1, W2P, (long long)4 * CDIM * CDIM, 48);

    patch_embed_kernel<<<dim3(BATCH * NTOK), 256, 0, stream>>>(x, patch_w, patch_b, pos_emb, h);

    for (int l = 0; l < D_LAYERS; ++l) {
        // qkv = h @ qkv_w[l]                  (128x64 block tiles)
        pack_a_bf16<<<dim3(12288 / 8, 1, 1), 256, 0, stream>>>(h, CDIM, 0, 0, 1, hA, 0, 24);
        gemm_packed_wmma<<<dim3(MROWS / 128, 36), 256, 0, stream>>>(
            hA, WqkvP + (size_t)l * CDIM * 3 * CDIM,
            nullptr, nullptr, qkv, 3 * CDIM, 24, 144, 0);

        // pack q (A- and B-side) and v (B-side) per (b,h)
        pack_a_bf16<<<dim3(48 / 8, 1, BH), 256, 0, stream>>>(
            qkv, 3 * CDIM, (long long)NTOK * 3 * CDIM, CH, HHEADS, qA, (long long)NTOK * CH, 3);
        pack_b_bf16<<<dim3(48 / 8, 1, BH), 256, 0, stream>>>(
            qkv, 3 * CDIM, (long long)NTOK * 3 * CDIM, CH, HHEADS, qB, (long long)NTOK * CH, 16);
        pack_b_bf16<<<dim3(48 / 8, 1, BH), 256, 0, stream>>>(
            qkv + 2 * CDIM, 3 * CDIM, (long long)NTOK * 3 * CDIM, CH, HHEADS, vB, (long long)NTOK * CH, 6);

        attn_scores_packed<<<dim3(16, 16, BH), 32, 0, stream>>>(qA, qB, S);
        softmax_mask_diag<<<dim3(BH * NTOK), 256, 0, stream>>>(S);
        laplacian_kernel<<<dim3(BH), 256, 0, stream>>>(S, lap);
        spectral_eigs_kernel<<<dim3(BH), 256, 0, stream>>>(lap, eig);
        group_attn_softmax<<<dim3(BH), 256, 0, stream>>>(eig, G);

        pack_a_bf16<<<dim3(16 / 8, 1, BH), 256, 0, stream>>>(
            G, NTOK, (long long)32 * NTOK, 0, 1, gAp, (long long)32 * NTOK, 8);
        group_out_packed<<<dim3(2, 6, BH), 32, 0, stream>>>(gAp, vB, aout);

        // h = aout @ proj_w[l] + proj_b[l]
        pack_a_bf16<<<dim3(1536 / 8, 1, 1), 256, 0, stream>>>(aout, CH, 0, 0, 1, aoutA, 0, 3);
        gemm_packed_wmma<<<dim3(MROWS / 128, 12), 256, 0, stream>>>(
            aoutA, WprojP + (size_t)l * CH * CDIM,
            proj_b + (size_t)l * CDIM, nullptr, h, CDIM, 3, 48, 0);

        layernorm_kernel<<<dim3(MROWS), 256, 0, stream>>>(
            h, g_ln_s + (size_t)l * CDIM, g_ln_b + (size_t)l * CDIM, hn);

        // mid = gelu(hn @ w1 + b1)
        pack_a_bf16<<<dim3(12288 / 8, 1, 1), 256, 0, stream>>>(hn, CDIM, 0, 0, 1, hnA, 0, 24);
        gemm_packed_wmma<<<dim3(MROWS / 128, 48), 256, 0, stream>>>(
            hnA, W1P + (size_t)l * CDIM * 4 * CDIM,
            mlp_b1 + (size_t)l * 4 * CDIM, nullptr, mid, 4 * CDIM, 24, 192, 1);

        // h = h + (mid @ w2 + b2)
        pack_a_bf16<<<dim3(49152 / 8, 1, 1), 256, 0, stream>>>(mid, 4 * CDIM, 0, 0, 1, midA, 0, 96);
        gemm_packed_wmma<<<dim3(MROWS / 128, 12), 256, 0, stream>>>(
            midA, W2P + (size_t)l * 4 * CDIM * CDIM,
            mlp_b2 + (size_t)l * CDIM, h, h, CDIM, 96, 48, 0);
    }

    head_kernel<<<dim3(BATCH), 256, 0, stream>>>(h, ln_s, ln_b, head_w, head_b, out);
}
